// BERTAttention_18691697672698
// MI455X (gfx1250) — compile-verified, matrix operand order checked by reading, not run on hardware
//
#include <hip/hip_runtime.h>
#include <hip/hip_bf16.h>

#define B_  4
#define S_  2048
#define H_  1024
#define NH_ 16
#define HD_ 64
#define EPS_ 1e-12f

typedef __attribute__((ext_vector_type(16))) __bf16 v16bf;
typedef __attribute__((ext_vector_type(8)))  __bf16 v8bf;
typedef __attribute__((ext_vector_type(8)))  float  v8f;
typedef __attribute__((ext_vector_type(4)))  unsigned int v4u;
typedef __attribute__((ext_vector_type(4)))  int    v4i;
typedef __attribute__((ext_vector_type(8)))  int    v8i;

#ifndef __has_builtin
#define __has_builtin(x) 0
#endif
#if __has_builtin(__builtin_amdgcn_tensor_load_to_lds)
#define USE_TDM 1
#else
#define USE_TDM 0
#endif

__device__ __forceinline__ unsigned short f32_to_bf16_rne(float f) {
  unsigned int u = __float_as_uint(f);
  u += 0x7FFFu + ((u >> 16) & 1u);
  return (unsigned short)(u >> 16);
}

// Load one 16x32 bf16 operand fragment (A-layout interleave; same loader used
// for B operands with "row" = output column, contiguous along K in memory).
__device__ __forceinline__ v16bf load_frag(const __bf16* base, int ld, int lane) {
  const int r = lane & 15;
  const int c = (lane >> 4) << 3;
  const __bf16* p = base + (size_t)r * ld + c;
  v8bf lo = *(const v8bf*)(p);
  v8bf hi = *(const v8bf*)(p + 16);
  v16bf f;
#pragma unroll
  for (int i = 0; i < 8; ++i) { f[i] = lo[i]; f[i + 8] = hi[i]; }
  return f;
}

__device__ __forceinline__ v8f wmma_bf16(v16bf a, v16bf b, v8f c) {
  return __builtin_amdgcn_wmma_f32_16x16x32_bf16(false, a, false, b, (short)0, c,
                                                 false, false);
}

// ------------------------------------------------------------------ TDM
// Issue a 2D TENSOR_LOAD_TO_LDS: tile_d1 rows x tile_d0 bf16 elements,
// global row stride = stride0 elements, destination ldsByteOff.
// D# packing per CDNA5 ISA ch.8: group0 = {count/type/addrs},
// group1 = {data_size, tensor dims, tile dims, strides}.
__device__ __forceinline__ void tdm_load_2d(unsigned ldsByteOff, const void* gaddr,
                                            unsigned tensor_d0, unsigned tensor_d1,
                                            unsigned tile_d0, unsigned tile_d1,
                                            unsigned stride0) {
#if USE_TDM
  const unsigned long long ga = (unsigned long long)gaddr;
  v4u g0;
  g0[0] = 1u;                                   // count=1, user descriptor
  g0[1] = ldsByteOff;                           // lds_addr [63:32]
  g0[2] = (unsigned)(ga & 0xFFFFFFFFu);         // global_addr lo
  g0[3] = (unsigned)((ga >> 32) & 0x01FFFFFFu)  // global_addr hi [120:96]
          | (2u << 30);                         // type=2 ("image")
  v8i g1;
  g1[0] = (int)(1u << 16);                      // data_size=1 -> 2 bytes
  g1[1] = (int)((tensor_d0 & 0xFFFFu) << 16);   // tensor_dim0 lo16 at [63:48]
  g1[2] = (int)((tensor_d0 >> 16) | ((tensor_d1 & 0xFFFFu) << 16));
  g1[3] = (int)((tensor_d1 >> 16) | (tile_d0 << 16));    // tile_dim0 [127:112]
  g1[4] = (int)(tile_d1 & 0xFFFFu);             // tile_dim1, tile_dim2=0
  g1[5] = (int)stride0;                         // tensor_dim0_stride lo32
  g1[6] = 0;                                    // stride0 hi16 | stride1 lo16
  g1[7] = 0;                                    // stride1 hi32
  v4i g2 = {0, 0, 0, 0};
  v4i g3 = {0, 0, 0, 0};
#if defined(__clang_major__) && (__clang_major__ >= 23)
  v8i g4 = {0, 0, 0, 0, 0, 0, 0, 0};
  __builtin_amdgcn_tensor_load_to_lds(g0, g1, g2, g3, g4, 0);
#else
  __builtin_amdgcn_tensor_load_to_lds(g0, g1, g2, g3, 0);
#endif
#else
  (void)ldsByteOff; (void)gaddr; (void)tensor_d0; (void)tensor_d1;
  (void)tile_d0; (void)tile_d1; (void)stride0;
#endif
}

// Stage a 16x64 bf16 slice of the weight matrix into LDS (one wave's share
// of the block's 64x64 double-buffered tile).
__device__ __forceinline__ void stage16x64(const __bf16* g, unsigned short* dst,
                                           unsigned ldsByteOff, int lane) {
#if USE_TDM
  (void)dst; (void)lane;
  tdm_load_2d(ldsByteOff, g, /*tensor_d0=*/H_, /*tensor_d1=*/1024,
              /*tile_d0=*/64, /*tile_d1=*/16, /*stride0=*/H_);
#else
  (void)ldsByteOff;
  const int r = lane >> 1;
  const int c = (lane & 1) * 32;
  const v8bf* s = (const v8bf*)(g + (size_t)r * H_ + c);
  v8bf* d = (v8bf*)(dst + r * 64 + c);
  d[0] = s[0]; d[1] = s[1]; d[2] = s[2]; d[3] = s[3];
#endif
}

// One 64-wide K chunk: A fragments streamed from global, B fragments read
// from the LDS-staged weight tile (wave's 32 output columns).
__device__ __forceinline__ void gemm_chunk(const __bf16* aR0, const __bf16* aR1,
                                           const unsigned short* swbuf, int nblk,
                                           int kc, int lane, v8f acc[2][2]) {
  const __bf16* wt = (const __bf16*)swbuf + (size_t)nblk * 32 * 64;
  __builtin_prefetch(aR0 + kc + 128, 0, 3);
  __builtin_prefetch(aR1 + kc + 128, 0, 3);
#pragma unroll
  for (int ks = 0; ks < 64; ks += 32) {
    v16bf a0 = load_frag(aR0 + kc + ks, H_, lane);
    v16bf a1 = load_frag(aR1 + kc + ks, H_, lane);
    v16bf b0 = load_frag(wt + ks, 64, lane);
    v16bf b1 = load_frag(wt + (size_t)16 * 64 + ks, 64, lane);
    acc[0][0] = wmma_bf16(a0, b0, acc[0][0]);
    acc[0][1] = wmma_bf16(a0, b1, acc[0][1]);
    acc[1][0] = wmma_bf16(a1, b0, acc[1][0]);
    acc[1][1] = wmma_bf16(a1, b1, acc[1][1]);
  }
}

// ---------------------------------------------------------------- convert
__global__ void cvt_f32_bf16(const float* __restrict__ in,
                             unsigned short* __restrict__ out, int n) {
  int i = blockIdx.x * blockDim.x + threadIdx.x;
  if (i < n) out[i] = f32_to_bf16_rne(in[i]);
}

// ---------------------------------------------------------------- QKV GEMM
// C = x @ W^T + b. blockDim=128 (4 waves); block tile 64(M)x64(N).
// Weight tile 64(N)x64(K) is TDM-staged into LDS, double buffered on
// TENSORcnt. z = 0/1/2 selects Q/K/V; V stored transposed [B,NH,HD,S].
__global__ void qkv_gemm(const __bf16* __restrict__ xb,
                         const __bf16* __restrict__ wqb,
                         const __bf16* __restrict__ wkb,
                         const __bf16* __restrict__ wvb,
                         const float* __restrict__ bq,
                         const float* __restrict__ bk,
                         const float* __restrict__ bv,
                         unsigned short* __restrict__ Qb,
                         unsigned short* __restrict__ Kb,
                         unsigned short* __restrict__ Vtb) {
  __shared__ __align__(128) unsigned short sw[2 * 64 * 64];  // 16 KB
  const int lane = threadIdx.x & 31;
  const int wave = __builtin_amdgcn_readfirstlane((int)(threadIdx.x >> 5));
  const int z = blockIdx.z;
  const __bf16* w   = (z == 0) ? wqb : (z == 1) ? wkb : wvb;
  const float* bias = (z == 0) ? bq  : (z == 1) ? bk  : bv;
  const int m0   = (blockIdx.x * 2 + (wave >> 1)) * 32;
  const int n0b  = blockIdx.y * 64;
  const int nblk = wave & 1;
  const int n0   = n0b + nblk * 32;

  v8f acc[2][2] = {};
  const __bf16* aR0 = xb + (size_t)m0 * H_;
  const __bf16* aR1 = xb + (size_t)(m0 + 16) * H_;
  const __bf16* wRow = w + (size_t)(n0b + wave * 16) * H_;
  const int sliceElems = wave * 1024;  // 16 rows * 64 cols

  stage16x64(wRow, sw + sliceElems, (unsigned)(sliceElems * 2), lane);
  int buf = 0;
  for (int kc = 0; kc < H_ - 64; kc += 64) {
    const int nb = buf ^ 1;
    stage16x64(wRow + kc + 64, sw + nb * 4096 + sliceElems,
               (unsigned)((nb * 4096 + sliceElems) * 2), lane);
    __builtin_amdgcn_s_wait_tensorcnt(1);
    __syncthreads();
    gemm_chunk(aR0, aR1, sw + buf * 4096, nblk, kc, lane, acc);
    __syncthreads();
    buf ^= 1;
  }
  __builtin_amdgcn_s_wait_tensorcnt(0);
  __syncthreads();
  gemm_chunk(aR0, aR1, sw + buf * 4096, nblk, H_ - 64, lane, acc);

  const int nl = lane & 15;
  const int mb = (lane >> 4) << 3;  // C layout: vgpr v holds M = mb + v
#pragma unroll
  for (int mi = 0; mi < 2; ++mi)
#pragma unroll
    for (int ni = 0; ni < 2; ++ni)
#pragma unroll
      for (int v = 0; v < 8; ++v) {
        const int m = m0 + mi * 16 + mb + v;   // = b*S + s
        const int n = n0 + ni * 16 + nl;       // = h*HD + d
        const float val = acc[mi][ni][v] + bias[n];
        const int bb = m >> 11;                // / S_
        const int s  = m & (S_ - 1);
        const int hh = n >> 6;
        const int d  = n & (HD_ - 1);
        const unsigned short o = f32_to_bf16_rne(val);
        if (z == 0)
          Qb[((size_t)(bb * NH_ + hh) * S_ + s) * HD_ + d] = o;
        else if (z == 1)
          Kb[((size_t)(bb * NH_ + hh) * S_ + s) * HD_ + d] = o;
        else
          Vtb[((size_t)(bb * NH_ + hh) * HD_ + d) * S_ + s] = o;
      }
}

// ---------------------------------------------------------------- attention
// One wave per 16 Q rows, flash-style online softmax over 32-key chunks.
// Per chunk: 4 WMMA (QK^T over HD=64) + 4 WMMA (PV over 4 feature tiles).
__global__ void flash_attn(const __bf16* __restrict__ Qb,
                           const __bf16* __restrict__ Kb,
                           const __bf16* __restrict__ Vtb,
                           unsigned short* __restrict__ ctxb) {
  __shared__ unsigned short ldsP[4][16 * 32];  // per-wave 16x32 P tile (bf16)
  const int lane = threadIdx.x & 31;
  const int wave = threadIdx.x >> 5;
  const int h = blockIdx.y;
  const int b = blockIdx.z;
  const int q0 = (blockIdx.x * 4 + wave) * 16;
  const __bf16* Q  = Qb  + (size_t)(b * NH_ + h) * S_ * HD_;
  const __bf16* K  = Kb  + (size_t)(b * NH_ + h) * S_ * HD_;
  const __bf16* Vt = Vtb + (size_t)(b * NH_ + h) * HD_ * S_;

  const v16bf qa0 = load_frag(Q + (size_t)q0 * HD_, HD_, lane);
  const v16bf qa1 = load_frag(Q + (size_t)q0 * HD_ + 32, HD_, lane);

  v8f acc[4] = {};
  float mrow[8], lrow[8], corr[8];
#pragma unroll
  for (int v = 0; v < 8; ++v) { mrow[v] = -3.0e38f; lrow[v] = 0.f; }

  unsigned short* lp = ldsP[wave];
  const int nl = lane & 15;
  const int mb = (lane >> 4) << 3;

  for (int kc = 0; kc < S_; kc += 32) {
    v8f s0 = {}, s1 = {};
    {
      v16bf kb0a = load_frag(K + (size_t)kc * HD_, HD_, lane);
      v16bf kb0b = load_frag(K + (size_t)kc * HD_ + 32, HD_, lane);
      v16bf kb1a = load_frag(K + (size_t)(kc + 16) * HD_, HD_, lane);
      v16bf kb1b = load_frag(K + (size_t)(kc + 16) * HD_ + 32, HD_, lane);
      s0 = wmma_bf16(qa0, kb0a, s0);
      s0 = wmma_bf16(qa1, kb0b, s0);
      s1 = wmma_bf16(qa0, kb1a, s1);
      s1 = wmma_bf16(qa1, kb1b, s1);
    }
    // Online softmax. Row m = mb+v lives in one 16-lane half at vgpr v;
    // xor masks 1,2,4,8 reduce across that half on wave32.
#pragma unroll
    for (int v = 0; v < 8; ++v) {
      float x0 = s0[v] * 0.125f;  // 1/sqrt(HD)
      float x1 = s1[v] * 0.125f;
      float cm = fmaxf(x0, x1);
      cm = fmaxf(cm, __shfl_xor(cm, 1, 32));
      cm = fmaxf(cm, __shfl_xor(cm, 2, 32));
      cm = fmaxf(cm, __shfl_xor(cm, 4, 32));
      cm = fmaxf(cm, __shfl_xor(cm, 8, 32));
      const float mn = fmaxf(mrow[v], cm);
      const float c  = __expf(mrow[v] - mn);
      const float p0 = __expf(x0 - mn);
      const float p1 = __expf(x1 - mn);
      float rs = p0 + p1;
      rs += __shfl_xor(rs, 1, 32);
      rs += __shfl_xor(rs, 2, 32);
      rs += __shfl_xor(rs, 4, 32);
      rs += __shfl_xor(rs, 8, 32);
      lrow[v] = lrow[v] * c + rs;
      mrow[v] = mn;
      corr[v] = c;
      const int row = mb + v;
      lp[row * 32 + nl]      = f32_to_bf16_rne(p0);  // keys kc..kc+15
      lp[row * 32 + 16 + nl] = f32_to_bf16_rne(p1);  // keys kc+16..kc+31
    }
#pragma unroll
    for (int f = 0; f < 4; ++f)
#pragma unroll
      for (int v = 0; v < 8; ++v) acc[f][v] *= corr[v];
    // P tile is per-wave private; a DS-counter wait suffices (no barrier).
    asm volatile("s_wait_dscnt 0" ::: "memory");
    const v16bf pa = load_frag((const __bf16*)lp, 32, lane);
#pragma unroll
    for (int f = 0; f < 4; ++f) {
      v16bf vb = load_frag(Vt + (size_t)(f * 16) * S_ + kc, S_, lane);
      acc[f] = wmma_bf16(pa, vb, acc[f]);
    }
  }
#pragma unroll
  for (int v = 0; v < 8; ++v) lrow[v] = 1.f / lrow[v];
#pragma unroll
  for (int f = 0; f < 4; ++f)
#pragma unroll
    for (int v = 0; v < 8; ++v) {
      const float o = acc[f][v] * lrow[v];
      ctxb[(size_t)(b * S_ + q0 + mb + v) * H_ + h * HD_ + f * 16 + nl] =
          f32_to_bf16_rne(o);
    }
}

// ---------------------------------------------------------------- out-proj
// h = ctx @ wo^T + bo + x  (fp32 out to workspace), TDM-staged weight tile.
__global__ void outproj_gemm(const __bf16* __restrict__ ab,
                             const __bf16* __restrict__ wb,
                             const float* __restrict__ bias,
                             const float* __restrict__ resid,
                             float* __restrict__ hbuf) {
  __shared__ __align__(128) unsigned short sw[2 * 64 * 64];  // 16 KB
  const int lane = threadIdx.x & 31;
  const int wave = __builtin_amdgcn_readfirstlane((int)(threadIdx.x >> 5));
  const int m0   = (blockIdx.x * 2 + (wave >> 1)) * 32;
  const int n0b  = blockIdx.y * 64;
  const int nblk = wave & 1;
  const int n0   = n0b + nblk * 32;

  v8f acc[2][2] = {};
  const __bf16* aR0 = ab + (size_t)m0 * H_;
  const __bf16* aR1 = ab + (size_t)(m0 + 16) * H_;
  const __bf16* wRow = wb + (size_t)(n0b + wave * 16) * H_;
  const int sliceElems = wave * 1024;

  stage16x64(wRow, sw + sliceElems, (unsigned)(sliceElems * 2), lane);
  int buf = 0;
  for (int kc = 0; kc < H_ - 64; kc += 64) {
    const int nb = buf ^ 1;
    stage16x64(wRow + kc + 64, sw + nb * 4096 + sliceElems,
               (unsigned)((nb * 4096 + sliceElems) * 2), lane);
    __builtin_amdgcn_s_wait_tensorcnt(1);
    __syncthreads();
    gemm_chunk(aR0, aR1, sw + buf * 4096, nblk, kc, lane, acc);
    __syncthreads();
    buf ^= 1;
  }
  __builtin_amdgcn_s_wait_tensorcnt(0);
  __syncthreads();
  gemm_chunk(aR0, aR1, sw + buf * 4096, nblk, H_ - 64, lane, acc);

  const int nl = lane & 15;
  const int mb = (lane >> 4) << 3;
#pragma unroll
  for (int mi = 0; mi < 2; ++mi)
#pragma unroll
    for (int ni = 0; ni < 2; ++ni)
#pragma unroll
      for (int v = 0; v < 8; ++v) {
        const int m = m0 + mi * 16 + mb + v;
        const int n = n0 + ni * 16 + nl;
        hbuf[(size_t)m * H_ + n] =
            acc[mi][ni][v] + bias[n] + resid[(size_t)m * H_ + n];
      }
}

// ---------------------------------------------------------------- layernorm
__global__ void layernorm_rows(const float* __restrict__ hbuf,
                               const float* __restrict__ gamma,
                               const float* __restrict__ beta,
                               float* __restrict__ out) {
  const int row = blockIdx.x;
  const float* hr = hbuf + (size_t)row * H_;
  float v[4];
  float sum = 0.f, sq = 0.f;
#pragma unroll
  for (int i = 0; i < 4; ++i) {
    v[i] = hr[threadIdx.x + i * 256];
    sum += v[i];
    sq  += v[i] * v[i];
  }
#pragma unroll
  for (int off = 16; off > 0; off >>= 1) {
    sum += __shfl_xor(sum, off, 32);
    sq  += __shfl_xor(sq,  off, 32);
  }
  __shared__ float ssum[8], ssq[8];
  const int wave = threadIdx.x >> 5, lane = threadIdx.x & 31;
  if (lane == 0) { ssum[wave] = sum; ssq[wave] = sq; }
  __syncthreads();
  if (wave == 0) {
    float s2 = (lane < 8) ? ssum[lane] : 0.f;
    float q2 = (lane < 8) ? ssq[lane]  : 0.f;
#pragma unroll
    for (int off = 4; off > 0; off >>= 1) {
      s2 += __shfl_xor(s2, off, 32);
      q2 += __shfl_xor(q2, off, 32);
    }
    if (lane == 0) { ssum[0] = s2; ssq[0] = q2; }
  }
  __syncthreads();
  const float mean = ssum[0] * (1.f / H_);
  const float var  = ssq[0] * (1.f / H_) - mean * mean;
  const float rstd = rsqrtf(var + EPS_);
#pragma unroll
  for (int i = 0; i < 4; ++i) {
    const int c = threadIdx.x + i * 256;
    out[(size_t)row * H_ + c] = gamma[c] * ((v[i] - mean) * rstd) + beta[c];
  }
}

// ---------------------------------------------------------------- launch
extern "C" void kernel_launch(void* const* d_in, const int* in_sizes, int n_in,
                              void* d_out, int out_size, void* d_ws,
                              size_t ws_size, hipStream_t stream) {
  (void)in_sizes; (void)n_in; (void)out_size; (void)ws_size;
  const float* x     = (const float*)d_in[0];
  const float* wq    = (const float*)d_in[1];
  const float* bq    = (const float*)d_in[2];
  const float* wk    = (const float*)d_in[3];
  const float* bk    = (const float*)d_in[4];
  const float* wv    = (const float*)d_in[5];
  const float* bv    = (const float*)d_in[6];
  const float* wo    = (const float*)d_in[7];
  const float* bo    = (const float*)d_in[8];
  const float* gamma = (const float*)d_in[9];
  const float* beta  = (const float*)d_in[10];
  float* out = (float*)d_out;

  char* ws = (char*)d_ws;
  const size_t NX = (size_t)B_ * S_ * H_;  // 8388608
  const size_t NW = (size_t)H_ * H_;       // 1048576
  size_t off = 0;
  unsigned short* xb   = (unsigned short*)(ws + off); off += NX * 2;
  unsigned short* wqb  = (unsigned short*)(ws + off); off += NW * 2;
  unsigned short* wkb  = (unsigned short*)(ws + off); off += NW * 2;
  unsigned short* wvb  = (unsigned short*)(ws + off); off += NW * 2;
  unsigned short* wob  = (unsigned short*)(ws + off); off += NW * 2;
  unsigned short* Qb   = (unsigned short*)(ws + off); off += NX * 2;
  unsigned short* Kb   = (unsigned short*)(ws + off); off += NX * 2;
  unsigned short* Vtb  = (unsigned short*)(ws + off); off += NX * 2;
  unsigned short* ctxb = (unsigned short*)(ws + off); off += NX * 2;
  float* hbuf          = (float*)(ws + off);          off += NX * 4;

  cvt_f32_bf16<<<(int)((NX + 255) / 256), 256, 0, stream>>>(x, xb, (int)NX);
  cvt_f32_bf16<<<(int)((NW + 255) / 256), 256, 0, stream>>>(wq, wqb, (int)NW);
  cvt_f32_bf16<<<(int)((NW + 255) / 256), 256, 0, stream>>>(wk, wkb, (int)NW);
  cvt_f32_bf16<<<(int)((NW + 255) / 256), 256, 0, stream>>>(wv, wvb, (int)NW);
  cvt_f32_bf16<<<(int)((NW + 255) / 256), 256, 0, stream>>>(wo, wob, (int)NW);

  qkv_gemm<<<dim3((B_ * S_) / 64, H_ / 64, 3), 128, 0, stream>>>(
      (const __bf16*)xb, (const __bf16*)wqb, (const __bf16*)wkb,
      (const __bf16*)wvb, bq, bk, bv, Qb, Kb, Vtb);

  flash_attn<<<dim3(S_ / 64, NH_, B_), 128, 0, stream>>>(
      (const __bf16*)Qb, (const __bf16*)Kb, (const __bf16*)Vtb, ctxb);

  outproj_gemm<<<dim3((B_ * S_) / 64, H_ / 64), 128, 0, stream>>>(
      (const __bf16*)ctxb, (const __bf16*)wob, bo, x, hbuf);

  layernorm_rows<<<dim3(B_ * S_), 256, 0, stream>>>(hbuf, gamma, beta, out);
}
